// NonLocalBlock_24240795418664
// MI455X (gfx1250) — matrix operand order repaired and checked by reading, not compiled, not run on hardware
//
#include <hip/hip_runtime.h>
#include <hip/hip_bf16.h>

// ---------------------------------------------------------------------------
// NonLocalBlock for MI455X (gfx1250, wave32, WMMA bf16 16x16x32)
// Flash-attention-style two-pass softmax over the query axis (dim=1),
// never materializing the 268MB f matrix. Pass 2 keeps the P tile entirely
// in registers: the transposed score tile's C-layout matches the B-operand
// layout of the x1 GEMM up to a lane<->lane^16 half swap (4 shfl_xor dwords).
// ---------------------------------------------------------------------------

typedef __bf16 v16bf __attribute__((ext_vector_type(16)));
typedef float  v8f   __attribute__((ext_vector_type(8)));

#define Bsz  4
#define Cdim 64
#define CiD  32
#define Npix 4096   // 64*64

union ABu { uint4 q[2]; unsigned int u[8]; v16bf v; };

__device__ __forceinline__ unsigned short f2bf(float x) {
  unsigned int u = __float_as_uint(x);
  u += 0x7fffu + ((u >> 16) & 1u);          // round-to-nearest-even
  return (unsigned short)(u >> 16);
}
__device__ __forceinline__ unsigned int pkbf(float lo, float hi) {
  return (unsigned int)f2bf(lo) | ((unsigned int)f2bf(hi) << 16);
}

// A-operand (M=16, K=32 bf16): A[M=row][K] = base[row*stride + k]
// lane<16: row=lane,    K{0..15}
// lane>=16: row=lane-16, K{16..31}
__device__ __forceinline__ v16bf load_a(const unsigned short* base,
                                        int rowstride, int lane) {
  const unsigned short* p = base + (size_t)(lane & 15) * rowstride + (lane >> 4) * 16;
  ABu u;
  u.q[0] = *(const uint4*)(p);
  u.q[1] = *(const uint4*)(p + 8);
  return u.v;
}

// B-operand (K=32, N=16) where B[k][n] = rows[n][k]; rows of 32 bf16 (64B).
// lane<16: col=lane, K0..15 (bytes 0..31); lane>=16: col=lane-16, K16..31.
__device__ __forceinline__ v16bf load_b_rows(const unsigned short* rowbase, int lane) {
  const char* p = (const char*)rowbase + (lane & 15) * 64 + (lane >> 4) * 32;
  ABu u;
  u.q[0] = *(const uint4*)(p);
  u.q[1] = *(const uint4*)(p + 16);
  return u.v;
}

// ---------------------------------------------------------------------------
// Kernel 1: 1x1 conv projections theta/phi/g (64->32), write bf16 workspace:
//   thetaT [b][n][c]  (n-major, 64B rows)  -> WMMA A/B operand (K=c)
//   phiT   [b][n][c]  (n-major)            -> WMMA A/B operand (K=c)
//   gC     [b][c][m]  (c-major)            -> WMMA A operand  (K=m)
// ---------------------------------------------------------------------------
__global__ void proj_kernel(const float* __restrict__ supp,
                            const float* __restrict__ ref,
                            const float* __restrict__ tw, const float* __restrict__ tb,
                            const float* __restrict__ pw, const float* __restrict__ pb,
                            const float* __restrict__ gw, const float* __restrict__ gb,
                            unsigned short* __restrict__ thetaT,
                            unsigned short* __restrict__ phiT,
                            unsigned short* __restrict__ gC) {
  __shared__ float wT[CiD * Cdim], wP[CiD * Cdim], wG[CiD * Cdim];
  __shared__ float bT[CiD], bP[CiD], bG[CiD];
  const int tid = threadIdx.x;
  for (int i = tid; i < CiD * Cdim; i += 256) { wT[i] = tw[i]; wP[i] = pw[i]; wG[i] = gw[i]; }
  if (tid < CiD) { bT[tid] = tb[tid]; bP[tid] = pb[tid]; bG[tid] = gb[tid]; }
  __syncthreads();

  const int pix = blockIdx.x * 256 + tid;       // pix = (b<<12) | n
  const int b   = pix >> 12;
  const int n   = pix & (Npix - 1);

  float aT[CiD], aP[CiD], aG[CiD];
#pragma unroll
  for (int co = 0; co < CiD; ++co) { aT[co] = bT[co]; aP[co] = bP[co]; aG[co] = bG[co]; }

  const float* sp = supp + ((size_t)b * Cdim) * Npix + n;
  const float* rp = ref  + ((size_t)b * Cdim) * Npix + n;
  for (int k = 0; k < Cdim; ++k) {
    const float s = sp[(size_t)k * Npix];
    const float r = rp[(size_t)k * Npix];
#pragma unroll
    for (int co = 0; co < CiD; ++co) {
      aT[co] = fmaf(wT[co * Cdim + k], s, aT[co]);
      aP[co] = fmaf(wP[co * Cdim + k], r, aP[co]);
      aG[co] = fmaf(wG[co * Cdim + k], r, aG[co]);
    }
  }

  unsigned short* trow = thetaT + (size_t)pix * CiD;
  unsigned short* prow = phiT   + (size_t)pix * CiD;
#pragma unroll
  for (int co = 0; co < CiD; ++co) {
    trow[co] = f2bf(aT[co]);
    prow[co] = f2bf(aP[co]);
    gC[((size_t)(b * CiD + co) << 12) + n] = f2bf(aG[co]);
  }
}

// ---------------------------------------------------------------------------
// Kernel 2: softmax stats over the QUERY axis n for each key column m.
// One wave owns 16 columns m; scans all 4096 n in 16-row WMMA tiles with an
// online (rescaling) max/sum. Lanes L and L^16 hold rows 0-7 / 8-15 of the
// same column -> combine with shfl_xor(16). Emits fused cAdj = max + log(sum)
// so pass 2 computes P = exp(f - cAdj) with a single subtract.
// ---------------------------------------------------------------------------
__global__ void stats_kernel(const unsigned short* __restrict__ thetaT,
                             const unsigned short* __restrict__ phiT,
                             float* __restrict__ cAdj) {
  const int lane = threadIdx.x & 31;
  const int wid  = (blockIdx.x * blockDim.x + threadIdx.x) >> 5;
  const int b    = wid >> 8;
  const int m0   = (wid & 255) << 4;

  const v16bf bphi = load_b_rows(phiT + ((size_t)((b << 12) + m0)) * CiD, lane);

  float runmax = -3.0e38f, runsum = 0.0f;
  for (int n0 = 0; n0 < Npix; n0 += 16) {
    v16bf a = load_a(thetaT + ((size_t)((b << 12) + n0)) * CiD, CiD, lane);
    if (n0 + 64 < Npix)
      __builtin_prefetch(thetaT + ((size_t)((b << 12) + n0 + 64)) * CiD, 0, 1);
    v8f c = {};
    c = __builtin_amdgcn_wmma_f32_16x16x32_bf16(false, a, false, bphi,
                                                (short)0, c, false, false);
    float tmax = c[0];
#pragma unroll
    for (int r = 1; r < 8; ++r) tmax = fmaxf(tmax, c[r]);
    tmax = fmaxf(tmax, __shfl_xor(tmax, 16, 32));
    const float nm = fmaxf(runmax, tmax);
    float t = 0.0f;
#pragma unroll
    for (int r = 0; r < 8; ++r) t += __expf(c[r] - nm);
    t += __shfl_xor(t, 16, 32);
    runsum = runsum * __expf(runmax - nm) + t;
    runmax = nm;
  }
  if (lane < 16)
    cAdj[(b << 12) + m0 + lane] = runmax + __logf(runsum);
}

// ---------------------------------------------------------------------------
// Kernel 3: attention pass, fully in registers.
// One wave owns a 16-column n tile. Per 32 m-rows:
//   fT[m,n] = A_phi x B_theta (two 16x16 WMMAs, K=c=32)
//   P = exp(fT - cAdj[m])  -> packed bf16; lane<->lane^16 half swap via
//   4 shfl_xor dwords yields the B-operand (K=m, N=n) directly.
//   x1T[c,n] += A_g(K=m) x B_P  (two WMMAs: c 0..15 and 16..31)
// ---------------------------------------------------------------------------
__global__ void attn_kernel(const unsigned short* __restrict__ thetaT,
                            const unsigned short* __restrict__ phiT,
                            const unsigned short* __restrict__ gC,
                            const float* __restrict__ cAdj,
                            float* __restrict__ x1C) {
  const int lane = threadIdx.x & 31;
  const int wid  = (blockIdx.x * blockDim.x + threadIdx.x) >> 5;
  const int b    = wid >> 8;
  const int n0   = (wid & 255) << 4;

  // B operand of the score GEMM: theta rows n0..n0+15, fixed for this wave.
  const v16bf btheta = load_b_rows(thetaT + ((size_t)((b << 12) + n0)) * CiD, lane);

  v8f d0 = {}, d1 = {};                 // x1T accumulators: c 0..15 / 16..31

  const int col   = lane & 15;          // C-layout column (n_local)
  const int rbase = (lane >> 4) * 8;    // C-layout row group (m_local 0 or 8)
  const bool lo   = (lane < 16);
  const float* ca = cAdj + (b << 12);

  for (int m0 = 0; m0 < Npix; m0 += 32) {
    float p0[8], p1[8];
#pragma unroll
    for (int h = 0; h < 2; ++h) {
      const int mh = m0 + h * 16;
      v16bf aphi = load_a(phiT + ((size_t)((b << 12) + mh)) * CiD, CiD, lane);
      v8f f = {};
      f = __builtin_amdgcn_wmma_f32_16x16x32_bf16(false, aphi, false, btheta,
                                                  (short)0, f, false, false);
      float* p = h ? p1 : p0;
#pragma unroll
      for (int r = 0; r < 8; ++r)
        p[r] = __expf(f[r] - ca[mh + rbase + r]);   // P[m,n] for this lane
    }
    // Pack pairs of consecutive m-rows (matches B-operand K-pair packing),
    // then swap complementary halves between lanes L and L^16.
    unsigned int own0[4], own1[4], send[4], recv[4];
#pragma unroll
    for (int j = 0; j < 4; ++j) {
      own0[j] = pkbf(p0[2 * j], p0[2 * j + 1]);
      own1[j] = pkbf(p1[2 * j], p1[2 * j + 1]);
      send[j] = lo ? own1[j] : own0[j];
    }
#pragma unroll
    for (int j = 0; j < 4; ++j)
      recv[j] = (unsigned int)__shfl_xor((int)send[j], 16, 32);
    ABu bp;
#pragma unroll
    for (int j = 0; j < 4; ++j) {
      bp.u[j]     = lo ? own0[j] : recv[j];   // K{0..7} | K{16..23}
      bp.u[4 + j] = lo ? recv[j] : own1[j];   // K{8..15}| K{24..31}
    }
    // A operands: g rows c (c-major, row stride 4096 bf16), K = m0..m0+31.
    v16bf ag0 = load_a(gC + ((size_t)(b * CiD)      << 12) + m0, Npix, lane);
    v16bf ag1 = load_a(gC + ((size_t)(b * CiD + 16) << 12) + m0, Npix, lane);

    d0 = __builtin_amdgcn_wmma_f32_16x16x32_bf16(false, ag0, false, bp.v,
                                                 (short)0, d0, false, false);
    d1 = __builtin_amdgcn_wmma_f32_16x16x32_bf16(false, ag1, false, bp.v,
                                                 (short)0, d1, false, false);
  }

  // D layout: col n = n0+col, rows c = rbase+r (d0) / 16+rbase+r (d1).
#pragma unroll
  for (int r = 0; r < 8; ++r) {
    const int c0 = rbase + r;
    x1C[((size_t)(b * CiD + c0)      << 12) + n0 + col] = d0[r];
    x1C[((size_t)(b * CiD + 16 + c0) << 12) + n0 + col] = d1[r];
  }
}

// ---------------------------------------------------------------------------
// Kernel 4: z = supp + W(x1) + b   (32 -> 64 1x1 conv, residual add)
// x1 is c-major [b][c][n]; loads are coalesced across lanes.
// ---------------------------------------------------------------------------
__global__ void out_kernel(const float* __restrict__ supp,
                           const float* __restrict__ x1C,
                           const float* __restrict__ ww, const float* __restrict__ wb,
                           float* __restrict__ z) {
  __shared__ float wS[Cdim * CiD];
  __shared__ float bS[Cdim];
  const int tid = threadIdx.x;
  for (int i = tid; i < Cdim * CiD; i += 256) wS[i] = ww[i];
  if (tid < Cdim) bS[tid] = wb[tid];
  __syncthreads();

  const int pix = blockIdx.x * 256 + tid;
  const int b   = pix >> 12;
  const int n   = pix & (Npix - 1);

  float xr[CiD];
#pragma unroll
  for (int ci = 0; ci < CiD; ++ci)
    xr[ci] = x1C[((size_t)(b * CiD + ci) << 12) + n];

  for (int co = 0; co < Cdim; ++co) {
    float acc = bS[co];
#pragma unroll
    for (int ci = 0; ci < CiD; ++ci) acc = fmaf(wS[co * CiD + ci], xr[ci], acc);
    const size_t off = ((size_t)(b * Cdim + co) << 12) + n;
    z[off] = supp[off] + acc;
  }
}

// ---------------------------------------------------------------------------
extern "C" void kernel_launch(void* const* d_in, const int* in_sizes, int n_in,
                              void* d_out, int out_size, void* d_ws, size_t ws_size,
                              hipStream_t stream) {
  const float* supp = (const float*)d_in[0];
  const float* ref  = (const float*)d_in[1];
  const float* tw   = (const float*)d_in[2];
  const float* tb   = (const float*)d_in[3];
  const float* pw   = (const float*)d_in[4];
  const float* pb   = (const float*)d_in[5];
  const float* gw   = (const float*)d_in[6];
  const float* gb   = (const float*)d_in[7];
  const float* ww   = (const float*)d_in[8];
  const float* wb   = (const float*)d_in[9];
  float* z = (float*)d_out;

  char* ws = (char*)d_ws;
  unsigned short* thetaT = (unsigned short*)(ws);                       // 1 MB
  unsigned short* phiT   = (unsigned short*)(ws + (1u << 20));          // 1 MB
  unsigned short* gC     = (unsigned short*)(ws + (2u << 20));          // 1 MB
  float* cAdj = (float*)(ws + (3u << 20));                              // 64 KB
  float* x1C  = (float*)(ws + (3u << 20) + (1u << 16));                 // 2 MB

  // 16384 pixels total; 256 threads/block
  proj_kernel <<<64,  256, 0, stream>>>(supp, ref, tw, tb, pw, pb, gw, gb,
                                        thetaT, phiT, gC);
  // 1024 waves = 4 batches * 256 m-tiles; 4 waves/block
  stats_kernel<<<256, 128, 0, stream>>>(thetaT, phiT, cAdj);
  // 1024 waves = 4 batches * 256 n-tiles; 4 waves/block
  attn_kernel <<<256, 128, 0, stream>>>(thetaT, phiT, gC, cAdj, x1C);
  out_kernel  <<<64,  256, 0, stream>>>(supp, x1C, ww, wb, z);
}
